// DetectionPostProcessor_12945031430229
// MI455X (gfx1250) — 8-bit matrix operand order repaired and checked by reading, compiled, not run on hardware
//
#include <hip/hip_runtime.h>
#include <hip/hip_bf16.h>
#include <stdint.h>

typedef __attribute__((ext_vector_type(16))) int   v16i;
typedef __attribute__((ext_vector_type(8)))  float v8f;

#define B_IMGS 32
#define TOPK   1000
#define KPAD   1024
#define DET    300
#define CAP    2048
#define BINS   4096
#define SCORE_THRESH 0.05f
#define NMS_THRESH   0.5f

#define HIST_OFF   0u
#define CNT_OFF    (HIST_OFF + B_IMGS*BINS)
#define CUT_OFF    (CNT_OFF + B_IMGS)
#define MASK_OFF   (CUT_OFF + B_IMGS)
#define MASK_WORDS (B_IMGS*KPAD*32)
#define ZERO_WORDS (MASK_OFF + MASK_WORDS)
#define COLL_OFF   ZERO_WORDS
#define SCORE_OFF  (COLL_OFF + B_IMGS*CAP*2)
#define LABEL_OFF  (SCORE_OFF + B_IMGS*KPAD)
#define AABB_OFF   (LABEL_OFF + B_IMGS*KPAD)
#define BOX_OFF    (AABB_OFF + B_IMGS*KPAD*4)

__global__ void k_zero(unsigned* __restrict__ p, int n) {
    int i = blockIdx.x * blockDim.x + threadIdx.x;
    if (i < n) p[i] = 0u;
}

__global__ void k_hist(const float* __restrict__ scores, unsigned* __restrict__ hist, int N) {
    __shared__ unsigned lh[BINS];
    int img = blockIdx.y;
    for (int i = threadIdx.x; i < BINS; i += blockDim.x) lh[i] = 0u;
    __syncthreads();
    const float* s = scores + (size_t)img * N;
    int stride = gridDim.x * blockDim.x;
    for (int i = blockIdx.x * blockDim.x + threadIdx.x; i < N; i += stride) {
        if (i + stride < N) __builtin_prefetch(&s[i + stride], 0, 1);
        float v = s[i];
        if (v > SCORE_THRESH) {
            int b = (int)(v * (float)BINS);
            if (b > BINS - 1) b = BINS - 1;
            if (b < 0) b = 0;
            atomicAdd(&lh[b], 1u);
        }
    }
    __syncthreads();
    unsigned* gh = hist + (size_t)img * BINS;
    for (int i = threadIdx.x; i < BINS; i += blockDim.x)
        if (lh[i]) atomicAdd(&gh[i], lh[i]);
}

__global__ void k_cut(const unsigned* __restrict__ hist, unsigned* __restrict__ cut) {
    int img = blockIdx.x;
    if (threadIdx.x == 0) {
        const unsigned* h = hist + (size_t)img * BINS;
        int lo = (int)(SCORE_THRESH * (float)BINS);
        unsigned acc = 0;
        int c = lo;
        for (int b = BINS - 1; b > lo; --b) {
            acc += h[b];
            if (acc >= TOPK) { c = b; break; }
        }
        cut[img] = (unsigned)c;
    }
}

__global__ void k_collect(const float* __restrict__ scores, const unsigned* __restrict__ cut,
                          unsigned* __restrict__ cnt, unsigned long long* __restrict__ coll, int N) {
    int img = blockIdx.y;
    const float* s = scores + (size_t)img * N;
    unsigned c = cut[img];
    int stride = gridDim.x * blockDim.x;
    for (int i = blockIdx.x * blockDim.x + threadIdx.x; i < N; i += stride) {
        if (i + stride < N) __builtin_prefetch(&s[i + stride], 0, 1);
        float v = s[i];
        if (v > SCORE_THRESH) {
            int b = (int)(v * (float)BINS);
            if (b > BINS - 1) b = BINS - 1;
            if ((unsigned)b >= c) {
                unsigned pos = atomicAdd(&cnt[img], 1u);
                if (pos < CAP) {
                    unsigned long long key =
                        ((unsigned long long)__float_as_uint(v) << 32) | (unsigned)(~(unsigned)i);
                    coll[(size_t)img * CAP + pos] = key;
                }
            }
        }
    }
}

__global__ void __launch_bounds__(1024) k_sort_gather(
    const unsigned long long* __restrict__ coll, const unsigned* __restrict__ cnt,
    const float* __restrict__ boxes, const int* __restrict__ labels, int N,
    float* __restrict__ sc_score, int* __restrict__ sc_label,
    float* __restrict__ sc_aabb, float* __restrict__ sc_box) {
    __shared__ unsigned long long keys[CAP];
    int img = blockIdx.x;
    int n = (int)cnt[img];
    if (n > CAP) n = CAP;
    const unsigned long long* c = coll + (size_t)img * CAP;
    for (int t = threadIdx.x; t < CAP; t += blockDim.x) keys[t] = (t < n) ? c[t] : 0ull;
    __syncthreads();
    for (int k = 2; k <= CAP; k <<= 1) {
        for (int j = k >> 1; j > 0; j >>= 1) {
            for (int t = threadIdx.x; t < CAP; t += blockDim.x) {
                int ixj = t ^ j;
                if (ixj > t) {
                    unsigned long long a = keys[t], b = keys[ixj];
                    bool up = ((t & k) == 0);
                    if (up ? (a < b) : (a > b)) { keys[t] = b; keys[ixj] = a; }
                }
            }
            __syncthreads();
        }
    }
    int t = threadIdx.x;
    unsigned long long key = keys[t];
    float sc = __uint_as_float((unsigned)(key >> 32));
    unsigned idx = ~(unsigned)key;
    bool valid = (t < TOPK) && (sc > SCORE_THRESH) && (idx < (unsigned)N);
    float b0 = 0, b1 = 0, b2 = 0, b3 = 0, b4 = 0;
    float x1 = 0, y1 = 0, x2 = 0, y2 = 0, scOut = 0;
    int lab = -1;
    if (valid) {
        const float* bp = boxes + ((size_t)img * N + idx) * 5;
        b0 = bp[0]; b1 = bp[1]; b2 = bp[2]; b3 = bp[3]; b4 = bp[4];
        lab = labels[(size_t)img * N + idx];
        float ca = fabsf(__cosf(b4)), sa = fabsf(__sinf(b4));
        float hx = 0.5f * (b2 * ca + b3 * sa);
        float hy = 0.5f * (b2 * sa + b3 * ca);
        x1 = b0 - hx; y1 = b1 - hy; x2 = b0 + hx; y2 = b1 + hy;
        scOut = sc;
    }
    size_t o = (size_t)img * KPAD + (size_t)t;
    sc_score[o] = scOut;
    sc_label[o] = lab;
    float* ab = sc_aabb + o * 4; ab[0] = x1; ab[1] = y1; ab[2] = x2; ab[3] = y2;
    float* bx = sc_box + o * 5;  bx[0] = b0; bx[1] = b1; bx[2] = b2; bx[3] = b3; bx[4] = b4;
}

__global__ void __launch_bounds__(32) k_mask(const int* __restrict__ sc_label,
                                             const float* __restrict__ sc_aabb,
                                             unsigned* __restrict__ mask) {
    int img = blockIdx.y;
    int tile = blockIdx.x;
    int ti = tile >> 6, tj = tile & 63;
    int L = (int)threadIdx.x;
    int hi = L >> 4;
    int m16 = L & 15;
    const int* lb = sc_label + (size_t)img * KPAD;
    int cm = lb[ti * 16 + m16];
    int cn = lb[tj * 16 + m16];
    v16i a, b;
#pragma unroll
    for (int v = 0; v < 16; ++v) {
        int v7 = v & 7;
        int kbA = ((v >> 3) << 6) + ((v7 >> 1) << 4) + ((v7 & 1) << 2) + (hi << 3);
        a[v] = (cm >= kbA && cm < kbA + 4) ? (int)(0x38u << ((cm - kbA) * 8)) : 0;
        int kbB = ((v >> 3) << 6) + ((v7 >> 1) << 4) + ((v7 & 1) << 2) + (hi << 3);
        b[v] = (cn >= kbB && cn < kbB + 4) ? (int)(0x38u << ((cn - kbB) * 8)) : 0;
    }
    v8f acc = {};
    acc = __builtin_amdgcn_wmma_f32_16x16x128_fp8_fp8(a, b, (short)0, acc, false, false);
    int gj = tj * 16 + m16;
    const float* abJ = sc_aabb + ((size_t)img * KPAD + gj) * 4;
    float jx1 = abJ[0], jy1 = abJ[1], jx2 = abJ[2], jy2 = abJ[3];
    float areaJ = (jx2 - jx1) * (jy2 - jy1);
    unsigned* mrow = mask + (size_t)img * KPAD * 32;
#pragma unroll
    for (int r = 0; r < 8; ++r) {
        int gi = ti * 16 + (hi << 3) + r;
        const float* abI = sc_aabb + ((size_t)img * KPAD + gi) * 4;
        float ix1 = abI[0], iy1 = abI[1], ix2 = abI[2], iy2 = abI[3];
        float areaI = (ix2 - ix1) * (iy2 - iy1);
        float iw = fmaxf(fminf(ix2, jx2) - fmaxf(ix1, jx1), 0.0f);
        float ih = fmaxf(fminf(iy2, jy2) - fmaxf(iy1, jy1), 0.0f);
        float inter = iw * ih;
        bool sup = (inter > NMS_THRESH * (areaI + areaJ - inter + 1e-7f)) && (acc[r] > 0.5f);
        unsigned bal = (unsigned)__ballot(sup);
        if (L == 0)
            atomicOr(&mrow[(size_t)(ti * 16 + r) * 32 + (tj >> 1)],
                     (bal & 0xFFFFu) << ((tj & 1) * 16));
        if (L == 16)
            atomicOr(&mrow[(size_t)(ti * 16 + 8 + r) * 32 + (tj >> 1)],
                     ((bal >> 16) & 0xFFFFu) << ((tj & 1) * 16));
    }
}

__global__ void k_merge(const float* __restrict__ sc_score, const int* __restrict__ sc_label,
                        const float* __restrict__ sc_box, const unsigned* __restrict__ mask,
                        float* __restrict__ out_b, float* __restrict__ out_s,
                        float* __restrict__ out_l) {
    __shared__ int keep[KPAD];
    __shared__ int rank[KPAD];
    int img = blockIdx.x;
    float* ob = out_b + (size_t)img * DET * 5;
    float* os = out_s + (size_t)img * DET;
    float* ol = out_l + (size_t)img * DET;
    for (int t = threadIdx.x; t < DET * 5; t += blockDim.x) ob[t] = 0.0f;
    for (int t = threadIdx.x; t < DET; t += blockDim.x) { os[t] = 0.0f; ol[t] = -1.0f; }
    __syncthreads();
    const float* ss = sc_score + (size_t)img * KPAD;
    if (threadIdx.x < 32) {
        int lane = (int)threadIdx.x;
        unsigned remv = 0u;
        const unsigned* m = mask + (size_t)img * KPAD * 32;
        for (int i = 0; i < KPAD; ++i) {
            unsigned sw = (unsigned)__shfl((int)remv, i >> 5, 32);
            bool alive = (ss[i] > SCORE_THRESH) && !((sw >> (i & 31)) & 1u);
            if (alive) remv |= m[(size_t)i * 32 + lane];
            if (lane == 0) keep[i] = alive ? 1 : 0;
        }
    }
    __syncthreads();
    if (threadIdx.x == 0) {
        int acc = 0;
        for (int i = 0; i < KPAD; ++i) { rank[i] = acc; acc += keep[i]; }
    }
    __syncthreads();
    const int* lb = sc_label + (size_t)img * KPAD;
    const float* bx = sc_box + (size_t)img * KPAD * 5;
    for (int t = threadIdx.x; t < KPAD; t += blockDim.x) {
        if (keep[t] && rank[t] < DET) {
            int p = rank[t];
            ob[p * 5 + 0] = bx[t * 5 + 0];
            ob[p * 5 + 1] = bx[t * 5 + 1];
            ob[p * 5 + 2] = bx[t * 5 + 2];
            ob[p * 5 + 3] = bx[t * 5 + 3];
            ob[p * 5 + 4] = bx[t * 5 + 4];
            os[p] = ss[t];
            ol[p] = (float)lb[t];
        }
    }
}

extern "C" void kernel_launch(void* const* d_in, const int* in_sizes, int n_in,
                              void* d_out, int out_size, void* d_ws, size_t ws_size,
                              hipStream_t stream) {
    const float* boxes  = (const float*)d_in[0];
    const float* scores = (const float*)d_in[1];
    const int*   labels = (const int*)d_in[2];
    int N = in_sizes[1] / B_IMGS;

    unsigned* ws = (unsigned*)d_ws;
    unsigned* hist = ws + HIST_OFF;
    unsigned* cnt  = ws + CNT_OFF;
    unsigned* cut  = ws + CUT_OFF;
    unsigned* mask = ws + MASK_OFF;
    unsigned long long* coll = (unsigned long long*)(ws + COLL_OFF);
    float* sc_score = (float*)(ws + SCORE_OFF);
    int*   sc_label = (int*)(ws + LABEL_OFF);
    float* sc_aabb  = (float*)(ws + AABB_OFF);
    float* sc_box   = (float*)(ws + BOX_OFF);

    float* out_b = (float*)d_out;
    float* out_s = out_b + (size_t)B_IMGS * DET * 5;
    float* out_l = out_s + (size_t)B_IMGS * DET;

    int zwords = (int)ZERO_WORDS;
    k_zero<<<dim3((zwords + 255) / 256), 256, 0, stream>>>(ws, zwords);
    k_hist<<<dim3(32, B_IMGS), 256, 0, stream>>>(scores, hist, N);
    k_cut<<<dim3(B_IMGS), 32, 0, stream>>>(hist, cut);
    k_collect<<<dim3(32, B_IMGS), 256, 0, stream>>>(scores, cut, cnt, coll, N);
    k_sort_gather<<<dim3(B_IMGS), 1024, 0, stream>>>(coll, cnt, boxes, labels, N,
                                                     sc_score, sc_label, sc_aabb, sc_box);
    k_mask<<<dim3(64 * 64, B_IMGS), 32, 0, stream>>>(sc_label, sc_aabb, mask);
    k_merge<<<dim3(B_IMGS), 256, 0, stream>>>(sc_score, sc_label, sc_box, mask,
                                              out_b, out_s, out_l);
}
